// CausalSelectiveSelfAttentionForInference_24713241821844
// MI455X (gfx1250) — compile-verified, matrix operand order checked by reading, not run on hardware
//
#include <hip/hip_runtime.h>

// ---------------- problem constants ----------------
#define TSEQ 2048
#define CDIM 1024
#define NH   16
#define HD   64
#define C3   3072          // 3*NH*HD
#define EFFK 408           // min cap: memory_budget-1 = int(2048*0.2)-1
#define NCH  16            // scan chunks
#define CHUNK (TSEQ / NCH) // 128 rows per chunk
#define MT   32            // GEMM macro-tile rows
#define NT   64            // GEMM macro-tile cols

typedef __attribute__((ext_vector_type(16))) __bf16 v16bf;
typedef __attribute__((ext_vector_type(8)))  __bf16 v8bf;
typedef __attribute__((ext_vector_type(2)))  __bf16 v2bf;
typedef __attribute__((ext_vector_type(8)))  float  v8f;

// fp32 -> bf16 via language-level fptrunc (backend picks native lowering; RTNE)
static __device__ __forceinline__ __bf16 f2bf(float f) {
  return (__bf16)f;
}
static __device__ __forceinline__ unsigned pack2bf(float lo, float hi) {
  v2bf p;
  p[0] = (__bf16)lo;
  p[1] = (__bf16)hi;
  return __builtin_bit_cast(unsigned, p);
}
static __device__ __forceinline__ v16bf cat8(v8bf lo, v8bf hi) {
  return __builtin_shufflevector(lo, hi, 0, 1, 2, 3, 4, 5, 6, 7, 8, 9, 10, 11, 12, 13, 14, 15);
}
static __device__ __forceinline__ int imin(int a, int b) { return a < b ? a : b; }

// ---- CDNA5 async global->LDS copy (ASYNCcnt-tracked) ----
static __device__ __forceinline__ void async_b128(unsigned lds_addr, const void* gaddr) {
  asm volatile("global_load_async_to_lds_b128 %0, %1, off"
               :: "v"(lds_addr), "v"((unsigned long long)(size_t)gaddr)
               : "memory");
}
static __device__ __forceinline__ void wait_async8() {
  asm volatile("s_wait_asynccnt 0x8" ::: "memory");
}
static __device__ __forceinline__ void wait_async0() {
  asm volatile("s_wait_asynccnt 0x0" ::: "memory");
}

// ============ shared GEMM body: bf16 WMMA with LDS staging of fp32 operands ============
// Computes Cmacro = Arows @ Brows^T for one 32x64 macro tile; epi(col, row_base, acc).
template <typename EPI>
static __device__ __forceinline__ void gemm_tile(const float* __restrict__ A,
                                                 const float* __restrict__ B,
                                                 int row0, int col0, EPI epi) {
  const int tid = threadIdx.x;
  const int wave = tid >> 5, lane = tid & 31;
  const int l15 = lane & 15, half = lane >> 4;
  const int my = (wave >> 2) * 16, nx = (wave & 3) * 16;

  __shared__ __align__(16) __bf16 As[MT * 32];
  __shared__ __align__(16) __bf16 Bs[NT * 32];

  const int sr = tid >> 3;            // 0..31 staging row
  const int sc = (tid & 7) * 4;       // 0,4,...,28 staging col

  v8f c;
  for (int r = 0; r < 8; ++r) c[r] = 0.0f;

  for (int kb = 0; kb < CDIM; kb += 32) {
    {
      const float4 av = *(const float4*)(A + (size_t)(row0 + sr) * CDIM + kb + sc);
      uint2 p; p.x = pack2bf(av.x, av.y); p.y = pack2bf(av.z, av.w);
      *(uint2*)&As[sr * 32 + sc] = p;
    }
    for (int it = 0; it < 2; ++it) {
      const int br = sr + it * 32;
      const float4 bv = *(const float4*)(B + (size_t)(col0 + br) * CDIM + kb + sc);
      uint2 p; p.x = pack2bf(bv.x, bv.y); p.y = pack2bf(bv.z, bv.w);
      *(uint2*)&Bs[br * 32 + sc] = p;
    }
    __syncthreads();
    const v8bf a_lo = *(const v8bf*)&As[(my + l15) * 32 + half * 8];
    const v8bf a_hi = *(const v8bf*)&As[(my + l15) * 32 + 16 + half * 8];
    const v8bf b_lo = *(const v8bf*)&Bs[(nx + l15) * 32 + half * 8];
    const v8bf b_hi = *(const v8bf*)&Bs[(nx + l15) * 32 + 16 + half * 8];
    c = __builtin_amdgcn_wmma_f32_16x16x32_bf16(false, cat8(a_lo, a_hi), false,
                                                cat8(b_lo, b_hi), (short)0, c, false, false);
    __syncthreads();
  }
  epi(col0 + nx + l15, row0 + my + (half << 3), c);
}

// ================= K1: QKV GEMM + scatter to head-major bf16 (V transposed) =================
__global__ void k1_qkv_gemm(const float* __restrict__ x, const float* __restrict__ Wa,
                            const float* __restrict__ ba,
                            __bf16* __restrict__ qbf, __bf16* __restrict__ kbf,
                            __bf16* __restrict__ vbfT) {
  const int nBx = C3 / NT;                 // 48
  const int bx = blockIdx.x % nBx, by = blockIdx.x / nBx;
  gemm_tile(x, Wa, by * MT, bx * NT, [&](int col, int row_base, v8f c) {
    const float bias = ba[col];
    const int sec = col >> 10;             // 0=q 1=k 2=v
    const int cc = col & 1023;
    const int hh = cc >> 6, dd = cc & 63;
    __bf16* base;
    size_t stride;
    if (sec == 2) { base = vbfT + (size_t)(hh * HD + dd) * TSEQ; stride = 1; }    // [h][d][T]
    else {
      base = (sec ? kbf : qbf) + (size_t)hh * TSEQ * HD + dd;
      stride = HD;
    }
    for (int r = 0; r < 8; ++r)
      base[(size_t)(row_base + r) * stride] = f2bf(c[r] + bias);
  });
}

// ================= K2: head-0 pruning scores S -> FF buffer (pre-scan) =================
__global__ void k2_head0_scores(const __bf16* __restrict__ qbf, const __bf16* __restrict__ kbf,
                                float* __restrict__ FF) {
  const int tile = blockIdx.x * 8 + (threadIdx.x >> 5);
  const int lane = threadIdx.x & 31;
  const int l15 = lane & 15, half = lane >> 4;
  const int nTx = TSEQ / 16;               // 128
  const int tx = tile % nTx, ty = tile / nTx;
  const int q0 = ty * 16, j0 = tx * 16;

  if (j0 >= q0 + 16) {                     // strictly above diagonal -> zeros (needs j < q)
    for (int r = 0; r < 8; ++r)
      FF[(size_t)(q0 + r + (half << 3)) * TSEQ + j0 + l15] = 0.0f;
    return;
  }
  const v8bf* qv = (const v8bf*)(qbf + (size_t)(q0 + l15) * HD);   // head 0
  const v8bf* kv = (const v8bf*)(kbf + (size_t)(j0 + l15) * HD);
  v8f c;
  for (int r = 0; r < 8; ++r) c[r] = 0.0f;
  c = __builtin_amdgcn_wmma_f32_16x16x32_bf16(false, cat8(qv[half], qv[2 + half]), false,
                                              cat8(kv[half], kv[2 + half]), (short)0, c, false, false);
  c = __builtin_amdgcn_wmma_f32_16x16x32_bf16(false, cat8(qv[4 + half], qv[6 + half]), false,
                                              cat8(kv[4 + half], kv[6 + half]), (short)0, c, false, false);
  for (int r = 0; r < 8; ++r) {
    const int q = q0 + r + (half << 3);
    const int j = j0 + l15;
    const float v = c[r] * 0.125f;         // 1/sqrt(64)
    FF[(size_t)q * TSEQ + j] = (j >= 1 && j < q) ? fmaxf(v, 0.0f) : 0.0f;
  }
}

// ================= K3: column-wise exclusive scan of FF (3 phases) =================
__global__ void k3a_colsum_part(const float* __restrict__ FF, float* __restrict__ part) {
  const int j = blockIdx.x * 256 + threadIdx.x;
  const int cchunk = blockIdx.y;
  float s = 0.0f;
  const int i0 = cchunk * CHUNK;
  for (int i = i0; i < i0 + CHUNK; ++i) s += FF[(size_t)i * TSEQ + j];
  part[(size_t)cchunk * TSEQ + j] = s;
}
__global__ void k3b_scan_part(float* __restrict__ part) {
  const int j = blockIdx.x * 256 + threadIdx.x;
  float run = 0.0f;
  for (int cchunk = 0; cchunk < NCH; ++cchunk) {
    const size_t idx = (size_t)cchunk * TSEQ + j;
    const float v = part[idx];
    part[idx] = run;
    run += v;
  }
}
__global__ void k3c_local_scan(float* __restrict__ FF, const float* __restrict__ part) {
  const int j = blockIdx.x * 256 + threadIdx.x;
  const int cchunk = blockIdx.y;
  float run = part[(size_t)cchunk * TSEQ + j];
  const int i0 = cchunk * CHUNK;
  for (int i = i0; i < i0 + CHUNK; ++i) {
    const size_t idx = (size_t)i * TSEQ + j;
    const float v = FF[idx];
    FF[idx] = run;                         // exclusive scan: FF[i] = sum S[0..i-1]
    run += v;
  }
}

// ================= K4: per-row k-th-smallest FF threshold (byte radix select) =================
__global__ void k4_select_rows(const float* __restrict__ FF, unsigned* __restrict__ sel) {
  const int i = blockIdx.x;
  const int tid = threadIdx.x;
  if (i < EFFK) {                          // eff_k == i+1 -> all causal keys kept
    if (tid == 0) { sel[i * 4 + 0] = 0xFFFFFFFFu; sel[i * 4 + 1] = 0; sel[i * 4 + 2] = 1u; }
    return;
  }
  const float* row = FF + (size_t)i * TSEQ;
  const int n = i + 1;                     // causal keys 0..i
  __shared__ unsigned hist[256];
  __shared__ unsigned eqc[256];
  __shared__ unsigned sh_prefix, sh_targ, sh_less;
  if (tid == 0) { sh_prefix = 0; sh_targ = EFFK - 1; sh_less = 0; }
  __syncthreads();
  for (int b = 3; b >= 0; --b) {
    hist[tid] = 0;
    __syncthreads();
    const unsigned prefix = sh_prefix;
    const unsigned hmask = (b == 3) ? 0u : (0xFFFFFFFFu << ((b + 1) * 8));
    for (int j = tid; j < n; j += 256) {
      const unsigned u = __float_as_uint(row[j]);        // FF >= 0 -> bits order-preserving
      if ((u & hmask) == (prefix & hmask))
        atomicAdd(&hist[(u >> (b * 8)) & 255u], 1u);
    }
    __syncthreads();
    if (tid == 0) {
      unsigned t = sh_targ, cum = 0;
      int bin = 0;
      for (; bin < 256; ++bin) { if (cum + hist[bin] > t) break; cum += hist[bin]; }
      sh_prefix = prefix | ((unsigned)bin << (b * 8));
      sh_targ = t - cum;
    }
    __syncthreads();
  }
  const unsigned thr = sh_prefix;          // exact bits of the rank-(EFFK-1) element
  unsigned cl = 0;
  for (int j = tid; j < n; j += 256)
    if (__float_as_uint(row[j]) < thr) cl++;
  atomicAdd(&sh_less, cl);
  const int chunk = (n + 255) / 256;
  const int js = tid * chunk, je = imin(n, js + chunk);
  unsigned eq = 0;
  for (int j = js; j < je; ++j)
    if (__float_as_uint(row[j]) == thr) eq++;
  eqc[tid] = eq;
  __syncthreads();
  if (tid == 0) {
    int rem = EFFK - (int)sh_less;         // >= 1
    int jcut = i;
    for (int t = 0; t < 256; ++t) {
      if ((int)eqc[t] >= rem) {
        int j = t * chunk;
        const int e = imin(n, j + chunk);
        for (; j < e; ++j)
          if (__float_as_uint(row[j]) == thr) { if (--rem == 0) { jcut = j; break; } }
        break;
      }
      rem -= (int)eqc[t];
    }
    sel[i * 4 + 0] = thr;
    sel[i * 4 + 1] = (unsigned)jcut;
    sel[i * 4 + 2] = 0u;
  }
}

// ================= K5: per-head flash attention with FF subtraction + selection =================
__global__ void k5_attention(const __bf16* __restrict__ qbf, const __bf16* __restrict__ kbf,
                             const __bf16* __restrict__ vbfT, const float* __restrict__ FF,
                             const unsigned* __restrict__ sel, float* __restrict__ yh) {
  const int qt = blockIdx.x, h = blockIdx.y;
  const int lane = threadIdx.x;
  const int l15 = lane & 15, half = lane >> 4;
  const int q0 = qt * 16;
  const int qmax = q0 + 15;

  __shared__ __align__(16) __bf16 Ks[32 * HD];      // 32 key rows x 64 dims
  __shared__ __align__(16) __bf16 Vs[HD * 32];      // 64 dim rows x 32 keys (V^T tile)
  __shared__ __align__(16) float  Ls[16 * 32];
  __shared__ __align__(16) __bf16 Ps[16 * 32];
  __shared__ float rscl[16], rsumS[16];

  // per-row state duplicated across lane pair (m, m+16): row m = l15
  float rmax_r = -3.0e38f, rsum_r = 0.0f;
  const int qi_row = q0 + l15;
  const unsigned thr = sel[qi_row * 4 + 0];
  const int jcut = (int)sel[qi_row * 4 + 1];
  const int keepall = (int)sel[qi_row * 4 + 2];
  const float* FFrow = FF + (size_t)qi_row * TSEQ;

  // Q tile in A layout (reduction dim = hd = 64 -> two K=32 operands)
  const v8bf* qv = (const v8bf*)(qbf + ((size_t)h * TSEQ + q0 + l15) * HD);
  const v16bf aq0 = cat8(qv[half], qv[2 + half]);
  const v16bf aq1 = cat8(qv[4 + half], qv[6 + half]);

  const __bf16* kb_base = kbf + (size_t)h * TSEQ * HD;
  const __bf16* vt_base = vbfT + (size_t)h * HD * TSEQ;
  const unsigned ks_lds = (unsigned)(size_t)&Ks[0];
  const unsigned vs_lds = (unsigned)(size_t)&Vs[0];

  v8f acc[4];
  for (int dt = 0; dt < 4; ++dt)
    for (int r = 0; r < 8; ++r) acc[dt][r] = 0.0f;

  for (int j0 = 0; j0 <= qmax; j0 += 32) {
    // --- async-stage K tile (8 x b128/lane-set), then V^T tile (8 more) ---
    for (int cc = 0; cc < 8; ++cc)         // lane owns key row j0+lane; 128B per row
      async_b128(ks_lds + (unsigned)(lane * 128 + cc * 16),
                 kb_base + (size_t)(j0 + lane) * HD + cc * 8);
    for (int s = 0; s < 2; ++s)            // lane owns dim row s*32+lane; 64B per row
      for (int cc = 0; cc < 4; ++cc)
        async_b128(vs_lds + (unsigned)((s * 32 + lane) * 64 + cc * 16),
                   vt_base + (size_t)(s * 32 + lane) * TSEQ + j0 + cc * 8);
    wait_async8();                         // K tile landed (async loads complete in order)

    // --- QK^T: two 16x16 logit subtiles from LDS ---
    for (int jt = 0; jt < 2; ++jt) {
      const v8bf* kv = (const v8bf*)&Ks[(jt * 16 + l15) * HD];
      v8f cq;
      for (int r = 0; r < 8; ++r) cq[r] = 0.0f;
      cq = __builtin_amdgcn_wmma_f32_16x16x32_bf16(false, aq0, false, cat8(kv[half], kv[2 + half]),
                                                   (short)0, cq, false, false);
      cq = __builtin_amdgcn_wmma_f32_16x16x32_bf16(false, aq1, false, cat8(kv[4 + half], kv[6 + half]),
                                                   (short)0, cq, false, false);
      for (int r = 0; r < 8; ++r)
        Ls[(r + (half << 3)) * 32 + jt * 16 + l15] = cq[r] * 0.125f;
    }
    __syncthreads();

    // --- online softmax: lane pair (m, m+16) splits row m's 32 keys ---
    {
      const int m = l15, jbase = half * 16;
      float tmax = -3.0e38f;
      for (int jj2 = 0; jj2 < 16; ++jj2) {
        const int jj = jbase + jj2;
        const int j = j0 + jj;
        float l = -3.0e38f;
        if (j <= qi_row) {
          const float ffv = FFrow[j];
          const unsigned fb = __float_as_uint(ffv);
          const bool keep = (j == qi_row) || keepall || (fb < thr) || (fb == thr && j <= jcut);
          if (keep) l = Ls[m * 32 + jj] - ffv;   // att = att - FF then mask
        }
        Ls[m * 32 + jj] = l;
        tmax = fmaxf(tmax, l);
      }
      tmax = fmaxf(tmax, __shfl_xor(tmax, 16, 32));
      const float nm = fmaxf(rmax_r, tmax);
      const float sc = __expf(rmax_r - nm);
      float ts = 0.0f;
      for (int jj2 = 0; jj2 < 16; ++jj2) {
        const int jj = jbase + jj2;
        const float p = __expf(Ls[m * 32 + jj] - nm);
        ts += p;
        Ps[m * 32 + jj] = f2bf(p);
      }
      ts += __shfl_xor(ts, 16, 32);
      rsum_r = rsum_r * sc + ts;
      rmax_r = nm;
      if (half == 0) rscl[m] = sc;
    }
    __syncthreads();

    // --- rescale accumulators, then P (16x32) @ V (32x64) via 4 WMMAs ---
    for (int dt = 0; dt < 4; ++dt)
      for (int r = 0; r < 8; ++r)
        acc[dt][r] *= rscl[r + (half << 3)];
    const v8bf* pv = (const v8bf*)&Ps[l15 * 32];
    const v16bf pa = cat8(pv[half], pv[2 + half]);     // P in A layout via LDS
    wait_async0();                         // V^T tile landed
    for (int dt = 0; dt < 4; ++dt) {
      const int d = dt * 16 + l15;
      const v8bf* vv = (const v8bf*)&Vs[d * 32];
      acc[dt] = __builtin_amdgcn_wmma_f32_16x16x32_bf16(false, pa, false, cat8(vv[half], vv[2 + half]),
                                                        (short)0, acc[dt], false, false);
    }
    __syncthreads();
  }

  if (half == 0) rsumS[l15] = rsum_r;
  __syncthreads();
  for (int dt = 0; dt < 4; ++dt) {
    const int d = dt * 16 + l15;
    for (int r = 0; r < 8; ++r) {
      const int m = r + (half << 3);
      yh[(size_t)(q0 + m) * CDIM + h * HD + d] = acc[dt][r] / rsumS[m];
    }
  }
}

// ================= K6: projection GEMM  out = yh @ W_proj^T + b_proj =================
__global__ void k6_proj_gemm(const float* __restrict__ yh, const float* __restrict__ Wp,
                             const float* __restrict__ bp, float* __restrict__ out) {
  const int nBx = CDIM / NT;               // 16
  const int bx = blockIdx.x % nBx, by = blockIdx.x / nBx;
  gemm_tile(yh, Wp, by * MT, bx * NT, [&](int col, int row_base, v8f c) {
    const float bias = bp[col];
    float* base = out + col;
    for (int r = 0; r < 8; ++r)
      base[(size_t)(row_base + r) * CDIM] = c[r] + bias;
  });
}

// ================================= launch =================================
extern "C" void kernel_launch(void* const* d_in, const int* in_sizes, int n_in,
                              void* d_out, int out_size, void* d_ws, size_t ws_size,
                              hipStream_t stream) {
  (void)in_sizes; (void)n_in; (void)out_size; (void)ws_size;
  const float* x  = (const float*)d_in[0];
  const float* Wa = (const float*)d_in[1];
  const float* ba = (const float*)d_in[2];
  const float* Wp = (const float*)d_in[3];
  const float* bp = (const float*)d_in[4];
  float* out = (float*)d_out;

  char* ws = (char*)d_ws;
  const size_t SZ_HB = (size_t)NH * TSEQ * HD * sizeof(unsigned short); // 4 MiB each
  __bf16*   qbf  = (__bf16*)(ws);
  __bf16*   kbf  = (__bf16*)(ws + SZ_HB);
  __bf16*   vbfT = (__bf16*)(ws + 2 * SZ_HB);                           // [h][d][T]
  float*    FF   = (float*)(ws + 3 * SZ_HB);                            // 16 MiB
  float*    yh   = (float*)(ws + 3 * SZ_HB + (size_t)TSEQ * TSEQ * 4);  // 8 MiB
  unsigned* sel  = (unsigned*)(ws + 3 * SZ_HB + (size_t)TSEQ * TSEQ * 4 + (size_t)TSEQ * CDIM * 4);
  float*    part = (float*)((char*)sel + (size_t)TSEQ * 4 * sizeof(unsigned));

  k1_qkv_gemm<<<(TSEQ / MT) * (C3 / NT), 256, 0, stream>>>(x, Wa, ba, qbf, kbf, vbfT);
  k2_head0_scores<<<(TSEQ / 16) * (TSEQ / 16) / 8, 256, 0, stream>>>(qbf, kbf, FF);
  k3a_colsum_part<<<dim3(TSEQ / 256, NCH), 256, 0, stream>>>(FF, part);
  k3b_scan_part<<<TSEQ / 256, 256, 0, stream>>>(part);
  k3c_local_scan<<<dim3(TSEQ / 256, NCH), 256, 0, stream>>>(FF, part);
  k4_select_rows<<<TSEQ, 256, 0, stream>>>(FF, sel);
  k5_attention<<<dim3(TSEQ / 16, NH), 32, 0, stream>>>(qbf, kbf, vbfT, FF, sel, yh);
  k6_proj_gemm<<<(TSEQ / MT) * (CDIM / NT), 256, 0, stream>>>(yh, Wp, bp, out);
}
